// Enc_Dec_Attention_21801253995051
// MI455X (gfx1250) — matrix operand order repaired and checked by reading, not compiled, not run on hardware
//
#include <hip/hip_runtime.h>
#include <hip/hip_bf16.h>

typedef __attribute__((ext_vector_type(16))) _Float16 v16h;
typedef __attribute__((ext_vector_type(8)))  _Float16 v8h;
typedef __attribute__((ext_vector_type(4)))  _Float16 v4h;
typedef __attribute__((ext_vector_type(8)))  float    v8f;

#define NXDIM 1024
#define NHEAD 16
#define HDIM  64
#define MROWS 8192   // B * T = 8 * 1024

union V16 { v16h v; v8h h[2]; };

// cross-half (lanes l <-> l^16) exchange via ds_swizzle SWAPX16
__device__ __forceinline__ float swap16f(float x) {
    return __int_as_float(__builtin_amdgcn_ds_swizzle(__float_as_int(x), 0x401f));
}

// CDNA5 async copy: 16 bytes global -> LDS, tracked by ASYNCcnt
__device__ __forceinline__ void async_g2l_b128(unsigned lds_off, const void* gp) {
    unsigned long long ga = (unsigned long long)(uintptr_t)gp;
    asm volatile("global_load_async_to_lds_b128 %0, %1, off"
                 :: "v"(lds_off), "v"(ga) : "memory");
}

// ---------------- fp32 -> fp16 bulk convert (x4 vectorized) ----------------
__global__ __launch_bounds__(256) void cvt_f32_f16_x4(const float* __restrict__ s,
                                                      _Float16* __restrict__ d, int n4) {
    int i = blockIdx.x * blockDim.x + threadIdx.x;
    if (i < n4) {
        float4 f = ((const float4*)s)[i];
        v4h o = { (_Float16)f.x, (_Float16)f.y, (_Float16)f.z, (_Float16)f.w };
        ((v4h*)d)[i] = o;
    }
}

// ---------------- bool mask -> float bias: masked -> -1e4, else +BIG ----------------
__global__ __launch_bounds__(256) void mask_to_f32(const unsigned char* __restrict__ m,
                                                   float* __restrict__ mf, int n) {
    int i = blockIdx.x * blockDim.x + threadIdx.x;
    if (i < n) mf[i] = m[i] ? -10000.0f : 3.0e38f;
}

// ---------------- fp32 [K,N] -> fp16 transposed [N,K] ----------------
__global__ __launch_bounds__(256) void transpose_cvt(const float* __restrict__ w,
                                                     _Float16* __restrict__ wt) {
    int id = blockIdx.x * blockDim.x + threadIdx.x;   // NX*NX threads
    int n = id & (NXDIM - 1);
    int k = id >> 10;
    wt[(size_t)n * NXDIM + k] = (_Float16)w[(size_t)k * NXDIM + n];
}

// ---------------- WMMA GEMM: C[M,NX] = A[M,NX](f16) @ W (via WT[n][k], f16) + bias ----
// MODE 0: f16 out to [B,H,T,D]   (q, k projections)
// MODE 1: f16 out to [B,H,D,T]   (v projection, pre-transposed for flash PV)
// MODE 2: f32 out to [M,NX]      (final c_proj)
template <int MODE>
__global__ __launch_bounds__(256) void gemm_wmma(const _Float16* __restrict__ A,
                                                 const _Float16* __restrict__ WT,
                                                 const float* __restrict__ bias,
                                                 void* __restrict__ outp) {
    const int lane = threadIdx.x & 31;
    const int lo = lane & 15, hi = lane >> 4;
    const int wave = threadIdx.x >> 5;
    const int task = blockIdx.x * 8 + wave;   // (M/16)*(NX/64) = 8192 wave tasks
    const int nt = task & 15;                 // 16 column tiles of 64
    const int mt = task >> 4;                 // 512 row tiles of 16

    const _Float16* Arow = A + (size_t)(mt * 16 + lo) * NXDIM;
    v8f acc[4] = {};

    #pragma unroll 1
    for (int kc = 0; kc < NXDIM; kc += 32) {
        // A fragment (16x32): lane lo = row; halves at K = hi*8 and 16+hi*8
        V16 a;
        a.h[0] = *(const v8h*)(Arow + kc + hi * 8);
        a.h[1] = *(const v8h*)(Arow + kc + 16 + hi * 8);
        __builtin_prefetch(Arow + kc + 64, 0, 1);
        // batch all four B fragments before the WMMAs
        V16 bf[4];
        #pragma unroll
        for (int sub = 0; sub < 4; ++sub) {
            const _Float16* Wrow = WT + (size_t)(nt * 64 + sub * 16 + lo) * NXDIM + kc + hi * 8;
            bf[sub].h[0] = *(const v8h*)(Wrow);
            bf[sub].h[1] = *(const v8h*)(Wrow + 16);
        }
        #pragma unroll
        for (int sub = 0; sub < 4; ++sub)
            acc[sub] = __builtin_amdgcn_wmma_f32_16x16x32_f16(
                false, a.v, false, bf[sub].v, (short)0, acc[sub], false, false);
    }

    const int mbase = mt * 16;
    const int b = mbase >> 10;       // 16 | 1024 so constant per tile
    const int trow = mbase & 1023;
    #pragma unroll
    for (int sub = 0; sub < 4; ++sub) {
        const int n = nt * 64 + sub * 16 + lo;
        const float bn = bias[n];
        if (MODE == 2) {
            float* out = (float*)outp;
            #pragma unroll
            for (int r = 0; r < 8; ++r) {
                int m = mbase + r + 8 * hi;
                out[(size_t)m * NXDIM + n] = acc[sub][r] + bn;
            }
        } else {
            _Float16* out = (_Float16*)outp;
            const int h = n >> 6, d = n & 63;
            #pragma unroll
            for (int r = 0; r < 8; ++r) {
                int t = trow + r + 8 * hi;
                float val = acc[sub][r] + bn;
                size_t idx;
                if (MODE == 0) idx = ((size_t)((b * NHEAD + h) * 1024) + t) * HDIM + d;
                else           idx = ((size_t)((b * NHEAD + h) * HDIM) + d) * 1024 + t;
                out[idx] = (_Float16)val;
            }
        }
    }
}

// ---------------- fused flash cross-attention, block-cooperative ----------------
// One block = one (b,h), 8 waves = 8 query tiles of 16.  K/V^T chunks (32 wide)
// are staged into LDS once per block with double-buffered async global->LDS DMA
// (GLOBAL_LOAD_ASYNC_TO_LDS_B128 + s_wait_asynccnt), overlapping DMA with WMMA.
__global__ __launch_bounds__(256) void flash_attn(const _Float16* __restrict__ qh,
                                                  const _Float16* __restrict__ kh,
                                                  const _Float16* __restrict__ vTh,
                                                  const float* __restrict__ maskf,
                                                  _Float16* __restrict__ ah) {
    __shared__ _Float16 Kst[2][32][64];   // [buf][k][d]   4 KB each
    __shared__ _Float16 Vst[2][64][32];   // [buf][d][k]   4 KB each
    __shared__ _Float16 P[8][16][32];     // per-wave 16q x 32k probability tile

    const int t    = threadIdx.x;
    const int lane = t & 31;
    const int lo = lane & 15, hi = lane >> 4;
    const int wave = t >> 5;
    const int qg = blockIdx.x & 7;            // query group of 8 tiles
    const int h  = (blockIdx.x >> 3) & 15;
    const int b  = blockIdx.x >> 7;           // 8*16*8 = 1024 blocks
    const int qbase = (qg * 8 + wave) * 16;

    const _Float16* Kbase = kh + (size_t)((b * NHEAD + h) * 1024) * HDIM;
    const _Float16* Vbase = vTh + (size_t)((b * NHEAD + h) * HDIM) * 1024;
    const float* mrow = maskf + (size_t)b * 1024;

    const unsigned kstLds = (unsigned)(uintptr_t)&Kst[0][0][0];  // low 32b = LDS offset
    const unsigned vstLds = (unsigned)(uintptr_t)&Vst[0][0][0];
    const int kr = t >> 3, kp = t & 7;        // K chunk: 32 rows x (8 x 16B)
    const int vd = t >> 2, vp = t & 3;        // V chunk: 64 rows x (4 x 16B)

    // Q as B-fragment of S^T: lane lo = query column; contiguous d-run of 16
    const _Float16* Qrow = qh + ((size_t)((b * NHEAD + h) * 1024) + qbase + lo) * HDIM;
    V16 qf0, qf1;
    qf0.h[0] = *(const v8h*)(Qrow + hi * 16);
    qf0.h[1] = *(const v8h*)(Qrow + hi * 16 + 8);
    qf1.h[0] = *(const v8h*)(Qrow + 32 + hi * 16);
    qf1.h[1] = *(const v8h*)(Qrow + 32 + hi * 16 + 8);

    float mrun = -3.0e38f, lsum = 0.0f;
    v8f o[4] = {};
    const float LOG2E = 1.44269504f;

    // prologue: stage chunk 0 into buffer 0 (2 async b128 per lane)
    async_g2l_b128(kstLds + kr * 128 + kp * 16, Kbase + (size_t)kr * HDIM + kp * 8);
    async_g2l_b128(vstLds + vd * 64 + vp * 16, Vbase + (size_t)vd * 1024 + vp * 8);

    #pragma unroll 1
    for (int it = 0; it < 32; ++it) {         // 1024 / 32 k-chunks
        const int kc = it * 32;
        const int cur = it & 1;
        if (it + 1 < 32) {                    // prefetch next chunk into other buffer
            const int nxt = (it + 1) & 1;
            const int kcn = kc + 32;
            async_g2l_b128(kstLds + nxt * 4096 + kr * 128 + kp * 16,
                           Kbase + (size_t)(kcn + kr) * HDIM + kp * 8);
            async_g2l_b128(vstLds + nxt * 4096 + vd * 64 + vp * 16,
                           Vbase + (size_t)vd * 1024 + kcn + vp * 8);
            asm volatile("s_wait_asynccnt 0x2" ::: "memory");   // chunk `it` landed
        } else {
            asm volatile("s_wait_asynccnt 0x0" ::: "memory");
        }
        __syncthreads();                      // all lanes' DMA for chunk `it` visible

        // ---- batch all 8 K-fragment DS loads, then the 4 S^T WMMAs ----
        V16 a[2][2];
        #pragma unroll
        for (int sub = 0; sub < 2; ++sub) {
            const _Float16* Krow = &Kst[cur][sub * 16 + lo][0];
            a[sub][0].h[0] = *(const v8h*)(Krow + hi * 16);
            a[sub][0].h[1] = *(const v8h*)(Krow + hi * 16 + 8);
            a[sub][1].h[0] = *(const v8h*)(Krow + 32 + hi * 16);
            a[sub][1].h[1] = *(const v8h*)(Krow + 32 + hi * 16 + 8);
        }
        v8f st[2];
        #pragma unroll
        for (int sub = 0; sub < 2; ++sub) {
            v8f c = {};
            c = __builtin_amdgcn_wmma_f32_16x16x32_f16(false, a[sub][0].v, false, qf0.v, (short)0, c, false, false);
            c = __builtin_amdgcn_wmma_f32_16x16x32_f16(false, a[sub][1].v, false, qf1.v, (short)0, c, false, false);
            st[sub] = c;
        }

        // ---- scale 1/8, mask via single v_min (bias is -1e4 or +BIG), row max ----
        float mloc = -3.0e38f;
        #pragma unroll
        for (int sub = 0; sub < 2; ++sub) {
            const float* mp = mrow + kc + sub * 16 + hi * 8;
            #pragma unroll
            for (int r = 0; r < 8; ++r) {
                float s = fminf(st[sub][r] * 0.125f, mp[r]);
                st[sub][r] = s;
                mloc = fmaxf(mloc, s);
            }
        }
        mloc = fmaxf(mloc, swap16f(mloc));          // combine lane halves
        float mnew = fmaxf(mrun, mloc);
        float corr = exp2f((mrun - mnew) * LOG2E);
        mrun = mnew;

        // ---- batch the 4 V fragments (8 DS loads) before the P round-trip ----
        V16 vf[4];
        #pragma unroll
        for (int dblk = 0; dblk < 4; ++dblk) {
            const _Float16* Vrow = &Vst[cur][dblk * 16 + lo][0];
            vf[dblk].h[0] = *(const v8h*)(Vrow + hi * 16);
            vf[dblk].h[1] = *(const v8h*)(Vrow + hi * 16 + 8);
        }

        // ---- exponentiate, row sum, stage P[q][k] in per-wave LDS tile ----
        float psum = 0.0f;
        #pragma unroll
        for (int sub = 0; sub < 2; ++sub) {
            v8h ph;
            #pragma unroll
            for (int r = 0; r < 8; ++r) {
                float p = exp2f((st[sub][r] - mnew) * LOG2E);
                psum += p;
                ph[r] = (_Float16)p;
            }
            *(v8h*)(&P[wave][lo][sub * 16 + hi * 8]) = ph;   // one b128 DS store
        }
        psum += swap16f(psum);
        lsum = lsum * corr + psum;

        asm volatile("s_wait_dscnt 0x0" ::: "memory");  // wave-private P tile

        // P^T as B-fragment: lane lo = query col, contiguous k-run of 16
        V16 pf;
        pf.h[0] = *(const v8h*)(&P[wave][lo][hi * 16]);
        pf.h[1] = *(const v8h*)(&P[wave][lo][hi * 16 + 8]);

        // ---- O^T += V^T . P^T over 4 d-blocks, flash rescale ----
        #pragma unroll
        for (int dblk = 0; dblk < 4; ++dblk) {
            #pragma unroll
            for (int r = 0; r < 8; ++r) o[dblk][r] *= corr;
            o[dblk] = __builtin_amdgcn_wmma_f32_16x16x32_f16(
                false, vf[dblk].v, false, pf.v, (short)0, o[dblk], false, false);
        }
        __syncthreads();   // buffer `cur` free for async overwrite next-next chunk
    }

    // ---- normalize and emit merged-head f16 rows (contiguous 16B stores) ----
    float rinv = 1.0f / lsum;
    _Float16* orow = ah + ((size_t)(b * 1024) + qbase + lo) * NXDIM + h * HDIM;
    #pragma unroll
    for (int dblk = 0; dblk < 4; ++dblk) {
        v8h oh;
        #pragma unroll
        for (int r = 0; r < 8; ++r) oh[r] = (_Float16)(o[dblk][r] * rinv);
        *(v8h*)(orow + dblk * 16 + hi * 8) = oh;
    }
}

// ---------------- host-side orchestration ----------------
extern "C" void kernel_launch(void* const* d_in, const int* in_sizes, int n_in,
                              void* d_out, int out_size, void* d_ws, size_t ws_size,
                              hipStream_t stream) {
    const float* x    = (const float*)d_in[0];
    const float* enc  = (const float*)d_in[1];
    const unsigned char* mask = (const unsigned char*)d_in[2];   // jnp bool -> 1 byte
    const float* wq = (const float*)d_in[3];
    const float* bq = (const float*)d_in[4];
    const float* wk = (const float*)d_in[5];
    const float* bk = (const float*)d_in[6];
    const float* wv = (const float*)d_in[7];
    const float* bv = (const float*)d_in[8];
    const float* wp = (const float*)d_in[9];
    const float* bp = (const float*)d_in[10];

    char* ws = (char*)d_ws;
    size_t off = 0;
    auto alloc = [&](size_t bytes) -> void* {
        void* p = ws + off;
        off += (bytes + 255) & ~(size_t)255;
        return p;
    };
    const size_t actsz = (size_t)MROWS * NXDIM * sizeof(_Float16);   // 16 MB
    const size_t wtsz  = (size_t)NXDIM * NXDIM * sizeof(_Float16);   //  2 MB
    _Float16* xh   = (_Float16*)alloc(actsz);
    _Float16* ench = (_Float16*)alloc(actsz);
    _Float16* wqT  = (_Float16*)alloc(wtsz);
    _Float16* wkT  = (_Float16*)alloc(wtsz);
    _Float16* wvT  = (_Float16*)alloc(wtsz);
    _Float16* wpT  = (_Float16*)alloc(wtsz);
    _Float16* qh   = (_Float16*)alloc(actsz);   // [B,H,Tq,D]
    _Float16* kh   = (_Float16*)alloc(actsz);   // [B,H,Tk,D]
    _Float16* vTh  = (_Float16*)alloc(actsz);   // [B,H,D,Tk]
    _Float16* ah   = (_Float16*)alloc(actsz);   // merged heads [M,NX]
    float*    maskf = (float*)alloc((size_t)8 * 1024 * sizeof(float));

    const int n4 = (MROWS * NXDIM) / 4;
    cvt_f32_f16_x4<<<n4 / 256, 256, 0, stream>>>(x,   xh,   n4);
    cvt_f32_f16_x4<<<n4 / 256, 256, 0, stream>>>(enc, ench, n4);
    mask_to_f32<<<(8 * 1024) / 256, 256, 0, stream>>>(mask, maskf, 8 * 1024);
    const int wthreads = NXDIM * NXDIM;
    transpose_cvt<<<wthreads / 256, 256, 0, stream>>>(wq, wqT);
    transpose_cvt<<<wthreads / 256, 256, 0, stream>>>(wk, wkT);
    transpose_cvt<<<wthreads / 256, 256, 0, stream>>>(wv, wvT);
    transpose_cvt<<<wthreads / 256, 256, 0, stream>>>(wp, wpT);

    // projections: 8192 wave tasks -> 1024 blocks x 8 waves
    gemm_wmma<0><<<1024, 256, 0, stream>>>(xh,   wqT, bq, qh);
    gemm_wmma<0><<<1024, 256, 0, stream>>>(ench, wkT, bk, kh);
    gemm_wmma<1><<<1024, 256, 0, stream>>>(ench, wvT, bv, vTh);

    // fused attention: block = (b,h) with 8 q-tiles; async double-buffered K/V
    flash_attn<<<1024, 256, 0, stream>>>(qh, kh, vTh, maskf, ah);

    // output projection straight to f32 d_out
    gemm_wmma<2><<<1024, 256, 0, stream>>>(ah, wpT, bp, d_out);
}
